// TopKSAE_85718957293620
// MI455X (gfx1250) — matrix operand order repaired and checked by reading, not compiled, not run on hardware
//
#include <hip/hip_runtime.h>

// ---------------------------------------------------------------------------
// TopK SAE forward for MI455X (gfx1250, wave32, WMMA).
//   h        = (x - b_dec) @ W_enc + b_enc          (2048 x 65536)  <- dominant
//   features = relu(h * topk_mask(h, 7))            (written in place over h)
//   recon    = features @ W_enc^T + b_dec           (2048 x 3584, 7-sparse)
// Encode GEMM: v_wmma_f32_16x16x32_bf16 (bf16 in, f32 accum), double-buffered
// LDS, B transposed into LDS with packed v_cvt_pk_bf16_f32 + b128 stores.
// ---------------------------------------------------------------------------

#define D_INP  3584
#define D_HIDE 65536
#define BATCH  2048
#define TOPK   7

#define BM 128            // block tile M (batch)
#define BN 128            // block tile N (hidden)
#define BK 32             // K step (matches wmma 16x16x32)
#define LDT 40            // LDS row pitch in halfwords (80 B: 16B-aligned, 20-dword stride)
#define NT (D_INP / BK)   // 112 K iterations

typedef __attribute__((ext_vector_type(16))) __bf16   v16bf;
typedef __attribute__((ext_vector_type(2)))  __bf16   bf16x2;
typedef __attribute__((ext_vector_type(8)))  float    v8f;
typedef __attribute__((ext_vector_type(4)))  unsigned u32x4;
typedef __attribute__((ext_vector_type(4)))  float    f32x4;
typedef __attribute__((ext_vector_type(2)))  float    f32x2;

union FragU { u32x4 q[2]; v16bf v; };

__device__ __forceinline__ unsigned pack_bf16(float lo, float hi) {
  bf16x2 p = { (__bf16)lo, (__bf16)hi };   // -> v_cvt_pk_bf16_f32
  union { bf16x2 v; unsigned u; } cv; cv.v = p;
  return cv.u;
}

// ---------------- K0: x' = bf16(x - b_dec), packed pairs -------------------
// grid (D_INP/512, BATCH): thread handles 2 consecutive cols of one row.
__global__ __launch_bounds__(256) void sae_prep_x(
    const float* __restrict__ x, const float* __restrict__ b_dec,
    unsigned* __restrict__ xp) {           // (BATCH, D_INP/2) packed bf16x2
  const int col2 = blockIdx.x * 256 + threadIdx.x;  // pair index within row
  const int row  = blockIdx.y;
  const f32x2 xv = *(const f32x2*)(x + (size_t)row * D_INP + col2 * 2);
  const f32x2 bd = *(const f32x2*)(b_dec + col2 * 2);
  xp[(size_t)row * (D_INP / 2) + col2] = pack_bf16(xv.x - bd.x, xv.y - bd.y);
}

// ---------------- K1: encode GEMM (bf16 WMMA, f32 accum) -------------------
__device__ __forceinline__ void load_tile(
    const unsigned short* __restrict__ xp, const float* __restrict__ W,
    unsigned short* As, unsigned short* Bs, int m0, int n0, int kt, int tid) {
  // A tile: 128x32 bf16 copied from precomputed x'; 2 threads per row,
  // 16 halfwords (32 B) each -> two ds_store_b128.
  const int a_m = tid >> 1;
  const int a_k = (tid & 1) << 4;
  const unsigned short* ap = xp + (size_t)(m0 + a_m) * D_INP + kt * BK + a_k;
  u32x4 av0 = *(const u32x4*)(ap);
  u32x4 av1 = *(const u32x4*)(ap + 8);
  *(u32x4*)(&As[a_m * LDT + a_k])     = av0;
  *(u32x4*)(&As[a_m * LDT + a_k + 8]) = av1;

  // B tile: 32x128 f32 from W_enc -> bf16, transposed into LDS as [n][k].
  // Each thread owns one n column and 16 of the 32 k's: global b32 loads are
  // fully coalesced across lanes (consecutive n); k-pairs packed into dwords
  // and written as two ds_store_b128 (k is the contiguous LDS axis).
  const int b_n  = tid & 127;
  const int b_k0 = (tid >> 7) << 4;     // 0 or 16
  const float* bp = W + (size_t)(kt * BK + b_k0) * D_HIDE + n0 + b_n;
  unsigned pk[8];
#pragma unroll
  for (int j = 0; j < 8; ++j) {
    float f0 = bp[(size_t)(2 * j) * D_HIDE];
    float f1 = bp[(size_t)(2 * j + 1) * D_HIDE];
    pk[j] = pack_bf16(f0, f1);
  }
  u32x4 q0 = { pk[0], pk[1], pk[2], pk[3] };
  u32x4 q1 = { pk[4], pk[5], pk[6], pk[7] };
  u32x4* dst = (u32x4*)(&Bs[b_n * LDT + b_k0]);
  dst[0] = q0;
  dst[1] = q1;
}

__global__ __launch_bounds__(256) void sae_encode_gemm(
    const unsigned short* __restrict__ xp,   // (BATCH, D_INP) bf16
    const float* __restrict__ W,             // (D_INP, D_HIDE) f32
    const float* __restrict__ b_enc,         // (D_HIDE)
    float* __restrict__ h) {                 // (BATCH, D_HIDE)
  __shared__ unsigned short As[2 * BM * LDT];
  __shared__ unsigned short Bs[2 * BN * LDT];

  const int tid  = threadIdx.x;
  const int lane = tid & 31;
  const int wv   = tid >> 5;      // 8 waves
  const int wM   = wv & 3;        // 4 waves along M -> 32 rows each
  const int wN   = wv >> 2;       // 2 waves along N -> 64 cols each
  const int m0   = blockIdx.x * BM;   // M fastest: 16 WGs share W panel in L2
  const int n0   = blockIdx.y * BN;

  v8f acc[2][4];
  const v8f vzero = {0.f, 0.f, 0.f, 0.f, 0.f, 0.f, 0.f, 0.f};
#pragma unroll
  for (int i = 0; i < 2; ++i)
#pragma unroll
    for (int j = 0; j < 4; ++j) acc[i][j] = vzero;

  load_tile(xp, W, As, Bs, m0, n0, 0, tid);

  int buf = 0;
  for (int kt = 0; kt < NT; ++kt) {
    __syncthreads();
    if (kt + 1 < NT)
      load_tile(xp, W, As + (buf ^ 1) * (BM * LDT), Bs + (buf ^ 1) * (BN * LDT),
                m0, n0, kt + 1, tid);

    const unsigned short* Ab = As + buf * (BM * LDT);
    const unsigned short* Bb = Bs + buf * (BN * LDT);

    // A frags: 16x32 bf16 layout — lane<16: rows M=lane, K {0..7,16..23};
    //          lane>=16: K {8..15,24..31}.
    FragU af[2];
    const int ar  = wM * 32 + (lane & 15);
    const int ak0 = (lane >> 4) << 3;
#pragma unroll
    for (int ms = 0; ms < 2; ++ms) {
      const unsigned short* p = Ab + (ar + ms * 16) * LDT + ak0;
      af[ms].q[0] = *(const u32x4*)(p);
      af[ms].q[1] = *(const u32x4*)(p + 16);
    }
    // B frags: 32x16 bf16 — lane<16: N=lane, K {0..7,16..23}; lane>=16: K {8..15,24..31}.
    FragU bfr[4];
    const int bc  = wN * 64 + (lane & 15);
    const int bk0 = (lane >> 4) << 3;
#pragma unroll
    for (int ns = 0; ns < 4; ++ns) {
      const unsigned short* p = Bb + (bc + ns * 16) * LDT + bk0;
      bfr[ns].q[0] = *(const u32x4*)(p);
      bfr[ns].q[1] = *(const u32x4*)(p + 16);
    }
#pragma unroll
    for (int ms = 0; ms < 2; ++ms)
#pragma unroll
      for (int ns = 0; ns < 4; ++ns)
        acc[ms][ns] = __builtin_amdgcn_wmma_f32_16x16x32_bf16(
            false, af[ms].v, false, bfr[ns].v, (short)0, acc[ms][ns],
            false, false);
    buf ^= 1;
  }

  // Epilogue: C/D layout — element r of v8f is row r + 8*(lane>=16), col lane%16.
#pragma unroll
  for (int ns = 0; ns < 4; ++ns) {
    const int n = n0 + wN * 64 + ns * 16 + (lane & 15);
    const float bias = b_enc[n];
#pragma unroll
    for (int ms = 0; ms < 2; ++ms) {
      const int r0 = m0 + wM * 32 + ms * 16 + ((lane >> 4) << 3);
#pragma unroll
      for (int r = 0; r < 8; ++r)
        h[(size_t)(r0 + r) * D_HIDE + n] = acc[ms][ns][r] + bias;
    }
  }
}

// ---------------- K2: per-row top-7 + relu mask (in place) -----------------
__global__ __launch_bounds__(256) void sae_topk_mask(
    float* __restrict__ h, float* __restrict__ topv, int* __restrict__ topi) {
  const int row = blockIdx.x;
  const int tid = threadIdx.x;
  float* hr = h + (size_t)row * D_HIDE;

  float tv[TOPK];
  int   ti[TOPK];
#pragma unroll
  for (int i = 0; i < TOPK; ++i) { tv[i] = -3.4e38f; ti[i] = 0; }

  for (int j = tid; j < D_HIDE; j += 256) {
    float v = hr[j];
    if (v > tv[TOPK - 1]) {
      tv[TOPK - 1] = v; ti[TOPK - 1] = j;
      int p = TOPK - 1;
      while (p > 0 && tv[p] > tv[p - 1]) {
        float fv = tv[p]; tv[p] = tv[p - 1]; tv[p - 1] = fv;
        int   iv = ti[p]; ti[p] = ti[p - 1]; ti[p - 1] = iv;
        --p;
      }
    }
  }

  __shared__ float sv[256 * TOPK];
  __shared__ int   si[256 * TOPK];
#pragma unroll
  for (int i = 0; i < TOPK; ++i) { sv[tid * TOPK + i] = tv[i]; si[tid * TOPK + i] = ti[i]; }
  __syncthreads();

  // tournament merge of sorted 7-lists
  for (int s = 128; s >= 1; s >>= 1) {
    if (tid < s) {
      float av[TOPK], bv[TOPK], mv[TOPK];
      int   ai[TOPK], bi[TOPK], mi[TOPK];
#pragma unroll
      for (int i = 0; i < TOPK; ++i) {
        av[i] = sv[tid * TOPK + i];        ai[i] = si[tid * TOPK + i];
        bv[i] = sv[(tid + s) * TOPK + i];  bi[i] = si[(tid + s) * TOPK + i];
      }
      int pa = 0, pb = 0;
#pragma unroll
      for (int i = 0; i < TOPK; ++i) {
        if (av[pa] >= bv[pb]) { mv[i] = av[pa]; mi[i] = ai[pa]; ++pa; }
        else                  { mv[i] = bv[pb]; mi[i] = bi[pb]; ++pb; }
      }
#pragma unroll
      for (int i = 0; i < TOPK; ++i) { sv[tid * TOPK + i] = mv[i]; si[tid * TOPK + i] = mi[i]; }
    }
    __syncthreads();
  }

  if (tid < TOPK) {
    topv[row * TOPK + tid] = sv[tid];
    topi[row * TOPK + tid] = si[tid];
  }
  const float thresh = sv[TOPK - 1];   // 7th largest

  for (int j = tid; j < D_HIDE; j += 256) {
    float v = hr[j];
    hr[j] = (v >= thresh && v > 0.f) ? v : 0.f;
  }
}

// ---------------- K3: sparse decode (7 columns/row) ------------------------
__global__ __launch_bounds__(256) void sae_decode(
    const float* __restrict__ topv, const int* __restrict__ topi,
    const float* __restrict__ W, const float* __restrict__ b_dec,
    float* __restrict__ recon) {
  const int row = blockIdx.x;
  const int tid = threadIdx.x;
  float fv[TOPK]; int fj[TOPK];
#pragma unroll
  for (int i = 0; i < TOPK; ++i) {
    float v = topv[row * TOPK + i];
    fv[i] = v > 0.f ? v : 0.f;          // relu'd feature value
    fj[i] = topi[row * TOPK + i];
  }
  for (int d = tid; d < D_INP; d += 256) {
    float acc = b_dec[d];
#pragma unroll
    for (int i = 0; i < TOPK; ++i)
      acc += fv[i] * W[(size_t)d * D_HIDE + fj[i]];
    recon[(size_t)row * D_INP + d] = acc;
  }
}

// ---------------------------------------------------------------------------
extern "C" void kernel_launch(void* const* d_in, const int* in_sizes, int n_in,
                              void* d_out, int out_size, void* d_ws, size_t ws_size,
                              hipStream_t stream) {
  (void)in_sizes; (void)n_in; (void)out_size; (void)ws_size;
  const float* x     = (const float*)d_in[0];   // (2048, 3584)
  const float* W     = (const float*)d_in[1];   // (3584, 65536)
  const float* b_enc = (const float*)d_in[2];   // (65536)
  const float* b_dec = (const float*)d_in[3];   // (3584)
  // d_in[4] = top_k scalar (compile-time TOPK=7)

  float* recon = (float*)d_out;                              // (2048, 3584)
  float* feats = (float*)d_out + (size_t)BATCH * D_INP;      // (2048, 65536)

  // workspace layout
  unsigned short* xp = (unsigned short*)d_ws;                          // 14,680,064 B
  char* wsb = (char*)d_ws;
  float* topv = (float*)(wsb + (size_t)BATCH * D_INP * 2);             // 57,344 B
  int*   topi = (int*)(wsb + (size_t)BATCH * D_INP * 2 + BATCH * TOPK * 4);

  dim3 gprep(D_INP / 512, BATCH);       // 2 cols per thread, 256 threads
  sae_prep_x<<<gprep, 256, 0, stream>>>(x, b_dec, (unsigned*)xp);

  dim3 grid(BATCH / BM, D_HIDE / BN);   // (16, 512); M fastest for L2 reuse of W
  sae_encode_gemm<<<grid, 256, 0, stream>>>(xp, W, b_enc, feats);

  sae_topk_mask<<<BATCH, 256, 0, stream>>>(feats, topv, topi);
  sae_decode<<<BATCH, 256, 0, stream>>>(topv, topi, W, b_dec, recon);
}
